// Decoder_50835232916319
// MI455X (gfx1250) — matrix operand order repaired and checked by reading, not compiled, not run on hardware
//
#include <hip/hip_runtime.h>
#include <math.h>

// ---- problem constants (match reference) ----
#define BB     32
#define TT     32
#define SS     400
#define EMB    512
#define ENC    512
#define DEC    512
#define ATT    512
#define VOCAB  50000
#define KPOOL  256      // DEC / POOL
#define G3     1536     // 3 * DEC
#define XK     1024     // EMB + ENC (GRU input-feed width)
#define RK     1536     // EMB + DEC + ENC (readout input width)

typedef __bf16 bf16;
typedef __attribute__((ext_vector_type(16))) __bf16 v16bf;
typedef __attribute__((ext_vector_type(8)))  __bf16 v8bf;
typedef __attribute__((ext_vector_type(8)))  float  v8f;

typedef __attribute__((ext_vector_type(4))) unsigned int u32x4;
typedef __attribute__((ext_vector_type(8))) int          i32x8;
typedef __attribute__((ext_vector_type(4))) int          i32x4;

union U16bf { v16bf v; v8bf h[2]; };

// ---------------- WMMA fragment helpers (gfx1250 16x16x32 bf16 layouts) ----
// A (16x32, row-major source [M,K]): lanes<16 rows 0..15 hold K {0..7,16..23},
// lanes 16..31 hold K {8..15,24..31} for the same rows.
__device__ __forceinline__ v16bf frag_a_bf16(const bf16* A, int lda, int m0, int k0) {
  int lane = threadIdx.x & 31;
  const bf16* p = A + (size_t)(m0 + (lane & 15)) * lda + k0;
  int klo = (lane < 16) ? 0 : 8;
  U16bf u;
  u.h[0] = *(const v8bf*)(p + klo);
  u.h[1] = *(const v8bf*)(p + 16 + klo);
  return u.v;
}

// Same A fragment but converting from an fp32 row-major source on the fly.
__device__ __forceinline__ v16bf frag_a_f32(const float* A, int lda, int m0, int k0) {
  int lane = threadIdx.x & 31;
  const float* p = A + (size_t)(m0 + (lane & 15)) * lda + k0 + ((lane < 16) ? 0 : 8);
  U16bf u;
#pragma unroll
  for (int i = 0; i < 8; ++i) u.h[0][i] = (bf16)p[i];
#pragma unroll
  for (int i = 0; i < 8; ++i) u.h[1][i] = (bf16)p[16 + i];
  return u.v;
}

// B (32x16) sourced from row-major weight W[N,K] (B = W^T):
// lane<16: N=lane, K = k0+{0..7,16..23} ; lane>=16: N=lane-16, K = k0+{8..15,24..31}.
__device__ __forceinline__ v16bf frag_b_w(const bf16* W, int ldw, int n0, int k0) {
  int lane = threadIdx.x & 31;
  const bf16* p = W + (size_t)(n0 + (lane & 15)) * ldw + k0 + ((lane < 16) ? 0 : 8);
  U16bf u;
  u.h[0] = *(const v8bf*)p;
  u.h[1] = *(const v8bf*)(p + 16);
  return u.v;
}

__device__ __forceinline__ v8f wmma_bf16(v16bf a, v16bf b, v8f c) {
  return __builtin_amdgcn_wmma_f32_16x16x32_bf16(false, a, false, b, (short)0, c,
                                                 false, false);
}

// ---------------- utility kernels ----------------
__global__ void k_cvt_bf16(const float* __restrict__ src, bf16* __restrict__ dst, int n) {
  int i = blockIdx.x * blockDim.x + threadIdx.x;
  if (i < n) dst[i] = (bf16)src[i];
}

__global__ void k_zero(float* __restrict__ p, int n) {
  int i = blockIdx.x * blockDim.x + threadIdx.x;
  if (i < n) p[i] = 0.0f;
}

// h0 = enc_states @ W_t^T + b_t   (one-time, fp32 VALU)
__global__ void k_h0(const float* __restrict__ enc_states, const float* __restrict__ Wt,
                     const float* __restrict__ bt, float* __restrict__ h) {
  int tid = blockIdx.x * blockDim.x + threadIdx.x;
  if (tid >= BB * DEC) return;
  int b = tid >> 9, d = tid & 511;
  const float* x = enc_states + b * ENC;
  const float* w = Wt + (size_t)d * ENC;
  float acc = bt[d];
  for (int e = 0; e < ENC; ++e) acc = fmaf(x[e], w[e], acc);
  h[tid] = acc;
}

// pre[b,s,:] = enc_out[b,s,:] @ W_pre^T + b_pre  -- M=12800, N=512, K=512 (WMMA)
__global__ void k_pre_gemm(const float* __restrict__ enc_out, const bf16* __restrict__ Wpre,
                           const float* __restrict__ b_pre, float* __restrict__ pre) {
  int wave = (blockIdx.x * blockDim.x + threadIdx.x) >> 5;
  const int ntilesN = ATT / 16;                       // 32
  const int tiles   = (BB * SS / 16) * ntilesN;       // 800*32
  if (wave >= tiles) return;
  int mt = wave / ntilesN, nt = wave % ntilesN;
  int lane = threadIdx.x & 31;
  v8f acc = {};
  for (int k0 = 0; k0 < ENC; k0 += 32) {
    v16bf a = frag_a_f32(enc_out, ENC, mt * 16, k0);
    v16bf b = frag_b_w(Wpre, ENC, nt * 16, k0);
    acc = wmma_bf16(a, b, acc);
  }
  int col = nt * 16 + (lane & 15);
  int rbase = mt * 16 + ((lane < 16) ? 0 : 8);
  float bv = b_pre[col];
#pragma unroll
  for (int v = 0; v < 8; ++v)
    pre[(size_t)(rbase + v) * ATT + col] = acc[v] + bv;
}

// Generic M=32 GEMM: out[32,N] = A_bf16[32,K] @ W_bf16[N,K]^T (+bias), fp32 out.
__global__ void k_gemm_m32(const bf16* __restrict__ A, int K,
                           const bf16* __restrict__ W,
                           const float* __restrict__ bias,
                           float* __restrict__ out, int ldo, int ntiles) {
  int wave = (blockIdx.x * blockDim.x + threadIdx.x) >> 5;
  if (wave >= ntiles) return;
  int lane = threadIdx.x & 31;
  int n0 = wave * 16;
  v8f acc0 = {}, acc1 = {};
  for (int k0 = 0; k0 < K; k0 += 32) {
    v16bf bf = frag_b_w(W, K, n0, k0);
    v16bf a0 = frag_a_bf16(A, K, 0, k0);
    v16bf a1 = frag_a_bf16(A, K, 16, k0);
    acc0 = wmma_bf16(a0, bf, acc0);
    acc1 = wmma_bf16(a1, bf, acc1);
  }
  int col = n0 + (lane & 15);
  float bv = bias ? bias[col] : 0.0f;
  int rbase = (lane < 16) ? 0 : 8;
#pragma unroll
  for (int v = 0; v < 8; ++v) {
    out[(size_t)(rbase + v) * ldo + col]      = acc0[v] + bv;
    out[(size_t)(16 + rbase + v) * ldo + col] = acc1[v] + bv;
  }
}

// ---------------- vocab logits GEMM: TDM -> LDS -> WMMA ----------------
// Each block stages a 128x256 bf16 slab of W_l into LDS via the Tensor Data
// Mover, then its 8 waves run the 16x16x32 WMMA K-loop out of LDS.
#define LROWS 128
#define LK    256     // == KPOOL

__global__ void __launch_bounds__(256)
k_logits_gemm(const bf16* __restrict__ A,      // maxout activations [32, 256] bf16
              const bf16* __restrict__ W,      // W_l bf16 [VOCAB, 256]
              const float* __restrict__ bias,  // b_l [VOCAB]
              float* __restrict__ out) {       // logits for this step [32, VOCAB]
  __shared__ bf16 tileB[LROWS * LK];           // 64 KB weight slab
  const int lane  = threadIdx.x & 31;
  const int wv    = threadIdx.x >> 5;          // 0..7
  const int n0blk = blockIdx.x * LROWS;

  if (wv == 0) {
    // ---- build 2D tensor descriptor (D#) and issue TDM load ----
    unsigned long long ga = (unsigned long long)(const void*)(W + (size_t)n0blk * LK);
    unsigned rem = (unsigned)(VOCAB - n0blk);
    if (rem > LROWS) rem = LROWS;              // OOB rows zero-fill
    u32x4 g0;
    g0[0] = 1u;                                 // count=1 (valid), user mode
    g0[1] = 0u;                                 // lds_addr = 0 (tileB at LDS base)
    g0[2] = (unsigned)(ga & 0xffffffffu);       // global_addr[31:0]
    g0[3] = (unsigned)((ga >> 32) & 0x1ffffffu) // global_addr[56:32]
            | (2u << 30);                       // type = 2 ("image")
    i32x8 g1;
    g1[0] = (1 << 16);                          // wg_mask=0, data_size=1 (2 bytes)
    g1[1] = (int)((LK & 0xffff) << 16);         // tensor_dim0[15:0]
    g1[2] = (int)(((unsigned)LK >> 16) & 0xffffu) | (int)((rem & 0xffffu) << 16);
    g1[3] = (int)((rem >> 16) & 0xffffu) | (LK << 16);   // tensor_dim1 hi | tile_dim0
    g1[4] = LROWS;                              // tile_dim1 | tile_dim2=0
    g1[5] = LK;                                 // tensor_dim0_stride[31:0]
    g1[6] = 0;                                  // stride0 hi | stride1 lo
    g1[7] = 0;                                  // stride1 hi
    i32x4 gz4 = {0, 0, 0, 0};                   // groups 2/3 unused (2D tensor)
    i32x8 gz8 = {0, 0, 0, 0, 0, 0, 0, 0};       // extra group (6-arg toolchain form)
    __builtin_amdgcn_tensor_load_to_lds(g0, g1, gz4, gz4, gz8, 0);
    __builtin_amdgcn_s_wait_tensorcnt(0);
  }
  __syncthreads();

  const int nl = wv * 16;                       // this wave's local column tile
  v8f acc0 = {}, acc1 = {};
#pragma unroll
  for (int k0 = 0; k0 < LK; k0 += 32) {
    // B fragment from LDS slab
    const bf16* p = &tileB[(size_t)(nl + (lane & 15)) * LK + k0 + ((lane < 16) ? 0 : 8)];
    U16bf ub;
    ub.h[0] = *(const v8bf*)p;
    ub.h[1] = *(const v8bf*)(p + 16);
    v16bf a0 = frag_a_bf16(A, LK, 0, k0);
    v16bf a1 = frag_a_bf16(A, LK, 16, k0);
    acc0 = wmma_bf16(a0, ub.v, acc0);
    acc1 = wmma_bf16(a1, ub.v, acc1);
  }
  int col = n0blk + nl + (lane & 15);
  if (n0blk + nl < VOCAB) {                     // tile-uniform guard (VOCAB % 16 == 0)
    float bv = bias[col];
    int rbase = (lane < 16) ? 0 : 8;
#pragma unroll
    for (int v = 0; v < 8; ++v) {
      out[(size_t)(rbase + v) * VOCAB + col]      = acc0[v] + bv;
      out[(size_t)(16 + rbase + v) * VOCAB + col] = acc1[v] + bv;
    }
  }
}

// Per-step prep: x_bf = [emb_t, ctx] (bf16), h_bf = cvt(h)
__global__ void k_prep_step(const float* __restrict__ trg, int t,
                            const float* __restrict__ ctx, const float* __restrict__ h,
                            bf16* __restrict__ xbf, bf16* __restrict__ hbf) {
  int tid = blockIdx.x * blockDim.x + threadIdx.x;
  if (tid < BB * XK) {
    int b = tid >> 10, j = tid & 1023;
    float v = (j < EMB) ? trg[((size_t)b * TT + t) * EMB + j] : ctx[b * ENC + (j - EMB)];
    xbf[tid] = (bf16)v;
  } else if (tid < BB * XK + BB * DEC) {
    int u = tid - BB * XK;
    hbf[u] = (bf16)h[u];
  }
}

// GRU gates -> h_new (writes h in place for next step) + bf16 copy for q GEMM
__global__ void k_gru_elem(const float* __restrict__ gi, const float* __restrict__ gh,
                           float* __restrict__ h, bf16* __restrict__ hnewbf) {
  int tid = blockIdx.x * blockDim.x + threadIdx.x;
  if (tid >= BB * DEC) return;
  int b = tid >> 9, d = tid & 511;
  const float* gib = gi + (size_t)b * G3;
  const float* ghb = gh + (size_t)b * G3;
  float ir = gib[d], iz = gib[d + DEC], in_ = gib[d + 2 * DEC];
  float hr = ghb[d], hz = ghb[d + DEC], hn = ghb[d + 2 * DEC];
  float r = 1.0f / (1.0f + expf(-(ir + hr)));
  float z = 1.0f / (1.0f + expf(-(iz + hz)));
  float n = tanhf(in_ + r * hn);
  float hnew = (1.0f - z) * n + z * h[tid];
  h[tid] = hnew;
  hnewbf[tid] = (bf16)hnew;
}

// energy[b,s] = sum_a tanh(pre + q + cov*w_cov) * v_att ; one wave per (b,s)
__global__ void k_energy(const float* __restrict__ pre, const float* __restrict__ q,
                         const float* __restrict__ cov, const float* __restrict__ w_cov,
                         const float* __restrict__ v_att, const int* __restrict__ mask,
                         float* __restrict__ energy, float* __restrict__ out_energy_t,
                         float* __restrict__ out_cov_t) {
  int wave = (blockIdx.x * blockDim.x + threadIdx.x) >> 5;
  if (wave >= BB * SS) return;
  int lane = threadIdx.x & 31;
  int b = wave / SS, s = wave % SS;
  float cv = cov[b * SS + s];
  const float* pr = pre + (size_t)(b * SS + s) * ATT;
  const float* qb = q + b * ATT;
  float partial = 0.0f;
#pragma unroll
  for (int i = 0; i < 16; ++i) {
    int a = lane + i * 32;
    partial += tanhf(pr[a] + qb[a] + cv * w_cov[a]) * v_att[a];
  }
#pragma unroll
  for (int off = 16; off > 0; off >>= 1)
    partial += __shfl_xor(partial, off, 32);
  if (lane == 0) {
    float e = (mask[b * SS + s] > 0) ? partial : -1.0e9f;
    energy[b * SS + s] = e;
    out_energy_t[b * SS + s] = e;
    out_cov_t[b * SS + s] = cv;     // coverage BEFORE update
  }
}

// softmax over S=400 per batch row; writes attn, attns output, cov += attn
__global__ void k_softmax(const float* __restrict__ energy, float* __restrict__ attn,
                          float* __restrict__ cov, float* __restrict__ out_attn_t) {
  __shared__ float red[512];
  int b = blockIdx.x, tid = threadIdx.x;
  float e = (tid < SS) ? energy[b * SS + tid] : -3.0e38f;
  red[tid] = e; __syncthreads();
  for (int s = 256; s > 0; s >>= 1) {
    if (tid < s) red[tid] = fmaxf(red[tid], red[tid + s]);
    __syncthreads();
  }
  float mx = red[0]; __syncthreads();
  float ex = (tid < SS) ? expf(e - mx) : 0.0f;
  red[tid] = ex; __syncthreads();
  for (int s = 256; s > 0; s >>= 1) {
    if (tid < s) red[tid] += red[tid + s];
    __syncthreads();
  }
  float inv = 1.0f / red[0];
  if (tid < SS) {
    float a = ex * inv;
    attn[b * SS + tid] = a;
    out_attn_t[b * SS + tid] = a;
    cov[b * SS + tid] += a;
  }
}

// context[b,d] = sum_s attn[b,s] * enc_out[b,s,d]
__global__ void k_context(const float* __restrict__ attn, const float* __restrict__ enc_out,
                          float* __restrict__ ctx) {
  int b = blockIdx.x, d = threadIdx.x;       // 512 threads
  const float* ab = attn + b * SS;
  const float* eb = enc_out + (size_t)b * SS * ENC + d;
  float acc = 0.0f;
  for (int s = 0; s < SS; ++s) acc = fmaf(ab[s], eb[(size_t)s * ENC], acc);
  ctx[b * ENC + d] = acc;
}

// copy_p[b] = sigmoid([h_new, ctx] . W_copy + b_copy)
__global__ void k_copy(const float* __restrict__ h, const float* __restrict__ ctx,
                       const float* __restrict__ Wc, const float* __restrict__ bc,
                       float* __restrict__ out_copy_t) {
  int b = threadIdx.x;
  if (b >= BB) return;
  float acc = bc[0];
  for (int j = 0; j < DEC; ++j) acc = fmaf(h[b * DEC + j], Wc[j], acc);
  for (int j = 0; j < ENC; ++j) acc = fmaf(ctx[b * ENC + j], Wc[DEC + j], acc);
  out_copy_t[b] = 1.0f / (1.0f + expf(-acc));
}

// readin = bf16([emb_t, h_new, ctx])
__global__ void k_prep_read(const float* __restrict__ trg, int t,
                            const float* __restrict__ h, const float* __restrict__ ctx,
                            bf16* __restrict__ readin) {
  int tid = blockIdx.x * blockDim.x + threadIdx.x;
  if (tid >= BB * RK) return;
  int b = tid / RK, j = tid % RK;
  float v;
  if (j < EMB)            v = trg[((size_t)b * TT + t) * EMB + j];
  else if (j < EMB + DEC) v = h[b * DEC + (j - EMB)];
  else                    v = ctx[b * ENC + (j - EMB - DEC)];
  readin[tid] = (bf16)v;
}

// maxout over adjacent pairs; emit bf16 for the vocab GEMM
__global__ void k_maxout(const float* __restrict__ readout, bf16* __restrict__ maxobf) {
  int tid = blockIdx.x * blockDim.x + threadIdx.x;
  if (tid >= BB * KPOOL) return;
  int b = tid >> 8, j = tid & 255;
  float m = fmaxf(readout[b * DEC + 2 * j], readout[b * DEC + 2 * j + 1]);
  maxobf[tid] = (bf16)m;
}

// ---------------- host-side orchestration ----------------
static inline size_t alignup(size_t x) { return (x + 255) & ~(size_t)255; }

extern "C" void kernel_launch(void* const* d_in, const int* in_sizes, int n_in,
                              void* d_out, int out_size, void* d_ws, size_t ws_size,
                              hipStream_t stream) {
  (void)in_sizes; (void)n_in; (void)out_size; (void)ws_size;
  const float* trg      = (const float*)d_in[0];
  // d_in[1] ext_src_seq: unused by reference math
  const float* enc_st   = (const float*)d_in[2];
  const float* enc_out  = (const float*)d_in[3];
  const int*   mask     = (const int*)  d_in[4];
  const float* W_t      = (const float*)d_in[5];
  const float* b_t      = (const float*)d_in[6];
  const float* W_ih     = (const float*)d_in[7];
  const float* W_hh     = (const float*)d_in[8];
  const float* b_ih     = (const float*)d_in[9];
  const float* b_hh     = (const float*)d_in[10];
  const float* W_pre    = (const float*)d_in[11];
  const float* b_pre    = (const float*)d_in[12];
  const float* W_q      = (const float*)d_in[13];
  const float* w_cov    = (const float*)d_in[14];
  const float* v_att    = (const float*)d_in[15];
  const float* W_copy   = (const float*)d_in[16];
  const float* b_copy   = (const float*)d_in[17];
  const float* W_r      = (const float*)d_in[18];
  const float* b_r      = (const float*)d_in[19];
  const float* W_l      = (const float*)d_in[20];
  const float* b_l      = (const float*)d_in[21];

  float* out        = (float*)d_out;
  float* out_logits = out;
  float* out_attn   = out + (size_t)TT * BB * VOCAB;
  float* out_cov    = out_attn + (size_t)TT * BB * SS;
  float* out_energy = out_cov + (size_t)TT * BB * SS;
  float* out_copy   = out_energy + (size_t)TT * BB * SS;

  // workspace carve-up
  char* w = (char*)d_ws;
  size_t o = 0;
  auto take = [&](size_t bytes) { char* p = w + o; o = alignup(o + bytes); return p; };
  bf16* Wl_bf   = (bf16*)take((size_t)VOCAB * KPOOL * 2);
  bf16* Wih_bf  = (bf16*)take((size_t)G3 * XK * 2);
  bf16* Whh_bf  = (bf16*)take((size_t)G3 * DEC * 2);
  bf16* Wpre_bf = (bf16*)take((size_t)ATT * ENC * 2);
  bf16* Wq_bf   = (bf16*)take((size_t)ATT * DEC * 2);
  bf16* Wr_bf   = (bf16*)take((size_t)DEC * RK * 2);
  float* pre    = (float*)take((size_t)BB * SS * ATT * 4);
  float* h      = (float*)take((size_t)BB * DEC * 4);
  float* ctx    = (float*)take((size_t)BB * ENC * 4);
  float* cov    = (float*)take((size_t)BB * SS * 4);
  bf16* xbf     = (bf16*)take((size_t)BB * XK * 2);
  bf16* hbf     = (bf16*)take((size_t)BB * DEC * 2);
  bf16* hnewbf  = (bf16*)take((size_t)BB * DEC * 2);
  float* gi     = (float*)take((size_t)BB * G3 * 4);
  float* gh     = (float*)take((size_t)BB * G3 * 4);
  float* q      = (float*)take((size_t)BB * ATT * 4);
  float* energy = (float*)take((size_t)BB * SS * 4);
  float* attn   = (float*)take((size_t)BB * SS * 4);
  bf16* readin  = (bf16*)take((size_t)BB * RK * 2);
  float* rdout  = (float*)take((size_t)BB * DEC * 4);
  bf16* maxobf  = (bf16*)take((size_t)BB * KPOOL * 2);

  auto cvt = [&](const float* src, bf16* dst, int n) {
    k_cvt_bf16<<<(n + 255) / 256, 256, 0, stream>>>(src, dst, n);
  };

  // one-time: weight conversions, zero state, h0, attention pre-compute
  cvt(W_l,  Wl_bf,  VOCAB * KPOOL);
  cvt(W_ih, Wih_bf, G3 * XK);
  cvt(W_hh, Whh_bf, G3 * DEC);
  cvt(W_pre, Wpre_bf, ATT * ENC);
  cvt(W_q,  Wq_bf,  ATT * DEC);
  cvt(W_r,  Wr_bf,  DEC * RK);
  k_zero<<<(BB * ENC + 255) / 256, 256, 0, stream>>>(ctx, BB * ENC);
  k_zero<<<(BB * SS + 255) / 256, 256, 0, stream>>>(cov, BB * SS);
  k_h0<<<(BB * DEC + 255) / 256, 256, 0, stream>>>(enc_st, W_t, b_t, h);
  {
    int tiles = (BB * SS / 16) * (ATT / 16);   // 25600 waves
    k_pre_gemm<<<(tiles + 7) / 8, 256, 0, stream>>>(enc_out, Wpre_bf, b_pre, pre);
  }

  const int nt_g3   = G3 / 16;                 // 96
  const int nt_512  = DEC / 16;                // 32
  const int nblk_v  = (VOCAB + LROWS - 1) / LROWS;  // 391 TDM blocks

  for (int t = 0; t < TT; ++t) {
    // 1. activations -> bf16 (x = [emb_t, ctx], h snapshot)
    k_prep_step<<<(BB * (XK + DEC) + 255) / 256, 256, 0, stream>>>(trg, t, ctx, h, xbf, hbf);
    // 2. GRU gate GEMMs (WMMA)
    k_gemm_m32<<<(nt_g3 + 7) / 8, 256, 0, stream>>>(xbf, XK,  Wih_bf, b_ih, gi, G3, nt_g3);
    k_gemm_m32<<<(nt_g3 + 7) / 8, 256, 0, stream>>>(hbf, DEC, Whh_bf, b_hh, gh, G3, nt_g3);
    // 3. gate nonlinearity -> h_new (in place)
    k_gru_elem<<<(BB * DEC + 255) / 256, 256, 0, stream>>>(gi, gh, h, hnewbf);
    // 4. q = h_new @ W_q^T (WMMA)
    k_gemm_m32<<<(nt_512 + 7) / 8, 256, 0, stream>>>(hnewbf, DEC, Wq_bf, nullptr, q, ATT, nt_512);
    // 5. coverage-attention energy (+ energies & pre-update coverage outputs)
    k_energy<<<(BB * SS + 7) / 8, 256, 0, stream>>>(pre, q, cov, w_cov, v_att, mask,
                                                    energy, out_energy + (size_t)t * BB * SS,
                                                    out_cov + (size_t)t * BB * SS);
    // 6. softmax + coverage update (+ attns output)
    k_softmax<<<BB, 512, 0, stream>>>(energy, attn, cov, out_attn + (size_t)t * BB * SS);
    // 7. context vector
    k_context<<<BB, 512, 0, stream>>>(attn, enc_out, ctx);
    // 8. copy gate
    k_copy<<<1, 32, 0, stream>>>(h, ctx, W_copy, b_copy, out_copy + (size_t)t * BB);
    // 9. readout input -> bf16
    k_prep_read<<<(BB * RK + 255) / 256, 256, 0, stream>>>(trg, t, h, ctx, readin);
    // 10. readout GEMM (WMMA)
    k_gemm_m32<<<(nt_512 + 7) / 8, 256, 0, stream>>>(readin, RK, Wr_bf, b_r, rdout, DEC, nt_512);
    // 11. maxout -> bf16
    k_maxout<<<(BB * KPOOL + 255) / 256, 256, 0, stream>>>(rdout, maxobf);
    // 12. vocab logits GEMM: TDM-staged LDS weights + WMMA (dominant cost)
    k_logits_gemm<<<nblk_v, 256, 0, stream>>>(maxobf, Wl_bf, b_l,
                                              out_logits + (size_t)t * BB * VOCAB);
  }
}
